// AttentionBlock3D_51702816309416
// MI455X (gfx1250) — matrix operand order repaired and checked by reading, not compiled, not run on hardware
//
#include <hip/hip_runtime.h>
#include <hip/hip_bf16.h>

// ---------------------------------------------------------------------------
// AttentionBlock3D for MI455X (gfx1250, wave32, WMMA)
// B=1, C=512, N=4096, HEADS=8, d=64, GROUPS=8
// All GEMMs + attention through v_wmma_f32_16x16x32_bf16 (f32 accumulate).
// Flash-attention tiling: never materializes the 8x4096x4096 score tensor.
// ---------------------------------------------------------------------------

typedef __attribute__((ext_vector_type(16))) __bf16 v16bf;
typedef __attribute__((ext_vector_type(8)))  float  v8f;

#define C_DIM   512
#define N_DIM   4096
#define HEADS   8
#define D_HEAD  64
#define GROUPS  8
#define GRP_ELEMS (64 * 4096)   // (C/GROUPS) * N

// ---------------- WMMA helpers (layouts per CDNA5 ISA 7.12.2) ---------------

__device__ __forceinline__ v8f wmma_bf16(v16bf a, v16bf b, v8f c) {
  // 8 args: (neg_a, A, neg_b, B, c_mod, C, reuse_a, reuse_b)
  return __builtin_amdgcn_wmma_f32_16x16x32_bf16(false, a, false, b, (short)0, c,
                                                 false, false);
}

// A-matrix 16(M)x32(K), 16-bit: lane = m (0..15) | half(k) in lane[4].
// VGPR j<8 -> k = j + 8*kh ; j>=8 -> k = j + 8 + 8*kh.
__device__ __forceinline__ v16bf frag_a(const __bf16* __restrict__ A, int lda,
                                        int m0, int k0) {
  int lane = threadIdx.x & 31;
  const __bf16* p = A + (size_t)(m0 + (lane & 15)) * lda + k0 + ((lane >> 4) << 3);
  v16bf a;
#pragma unroll
  for (int j = 0; j < 8; ++j) a[j] = p[j];
#pragma unroll
  for (int j = 8; j < 16; ++j) a[j] = p[j + 8];
  return a;
}

// B-matrix 32(K)x16(N), 16-bit: lane = n (0..15) | half(k) in lane[4].
// VGPR j<8 -> k = j + 8*kh ; j>=8 -> k = j + 8 + 8*kh.
__device__ __forceinline__ v16bf frag_b(const __bf16* __restrict__ B, int ldb,
                                        int k0, int n0) {
  int lane = threadIdx.x & 31;
  const __bf16* p = B + (size_t)(k0 + ((lane >> 4) << 3)) * ldb + n0 + (lane & 15);
  v16bf b;
#pragma unroll
  for (int j = 0; j < 8; ++j) b[j] = p[(size_t)j * ldb];
#pragma unroll
  for (int j = 8; j < 16; ++j) b[j] = p[(size_t)(j + 8) * ldb];
  return b;
}
// C/D 16x16 f32: VGPR r, lanes 0-15 -> (M=r, N=lane); lanes 16-31 -> (M=r+8).

// ---------------------- GroupNorm: two-stage reduction ----------------------

__global__ __launch_bounds__(256) void gn_partial(const float* __restrict__ x,
                                                  float* __restrict__ partials) {
  int g = blockIdx.x >> 4, slice = blockIdx.x & 15;   // 16 blocks per group
  const float4* xg = (const float4*)(x + (size_t)g * GRP_ELEMS) + slice * 4096;
  float s = 0.f, ss = 0.f;
  for (int i = threadIdx.x; i < 4096; i += 256) {
    float4 v = xg[i];
    s  += v.x + v.y + v.z + v.w;
    ss += v.x * v.x + v.y * v.y + v.z * v.z + v.w * v.w;
  }
  __shared__ float sh[512];
  sh[threadIdx.x] = s;
  sh[256 + threadIdx.x] = ss;
  __syncthreads();
  for (int off = 128; off > 0; off >>= 1) {
    if ((int)threadIdx.x < off) {
      sh[threadIdx.x]       += sh[threadIdx.x + off];
      sh[256 + threadIdx.x] += sh[256 + threadIdx.x + off];
    }
    __syncthreads();
  }
  if (threadIdx.x == 0) {
    partials[blockIdx.x * 2 + 0] = sh[0];
    partials[blockIdx.x * 2 + 1] = sh[256];
  }
}

__global__ void gn_finalize(const float* __restrict__ partials,
                            float* __restrict__ stats) {
  int g = threadIdx.x;
  if (g < GROUPS) {
    float s = 0.f, ss = 0.f;
    for (int i = 0; i < 16; ++i) {
      s  += partials[(g * 16 + i) * 2 + 0];
      ss += partials[(g * 16 + i) * 2 + 1];
    }
    const float inv = 1.0f / (float)GRP_ELEMS;
    float mean = s * inv;
    float var  = ss * inv - mean * mean;
    stats[g * 2 + 0] = mean;
    stats[g * 2 + 1] = rsqrtf(var + 1e-5f);
  }
}

__global__ __launch_bounds__(256) void gn_apply(const float* __restrict__ x,
                                                const float* __restrict__ w,
                                                const float* __restrict__ b,
                                                const float* __restrict__ stats,
                                                __bf16* __restrict__ h) {
  int i4 = blockIdx.x * 256 + threadIdx.x;  // over 524288 float4
  int c  = i4 >> 10;                        // 1024 float4 per channel
  int g  = c >> 6;
  float mean = stats[g * 2 + 0], rstd = stats[g * 2 + 1];
  float sc = w[c] * rstd, bb = b[c] - mean * sc;
  float4 v = ((const float4*)x)[i4];
  __bf16* p = h + (size_t)i4 * 4;
  p[0] = (__bf16)(v.x * sc + bb);
  p[1] = (__bf16)(v.y * sc + bb);
  p[2] = (__bf16)(v.z * sc + bb);
  p[3] = (__bf16)(v.w * sc + bb);
}

__global__ __launch_bounds__(256) void cvt_bf16_k(const float* __restrict__ s,
                                                  __bf16* __restrict__ d, int n4) {
  int i = blockIdx.x * 256 + threadIdx.x;
  if (i < n4) {
    float4 v = ((const float4*)s)[i];
    __bf16* p = d + (size_t)i * 4;
    p[0] = (__bf16)v.x; p[1] = (__bf16)v.y; p[2] = (__bf16)v.z; p[3] = (__bf16)v.w;
  }
}

// -------- QKV GEMM: [1536,512] x [512,4096], scatter into attn layouts ------
// block = 128(M) x 128(N), 8 waves (4x2), each wave 32x64 (2x4 C tiles).

__global__ __launch_bounds__(256) void qkv_gemm(const __bf16* __restrict__ W,
                                                const __bf16* __restrict__ H,
                                                const float* __restrict__ bias,
                                                __bf16* __restrict__ qb,
                                                __bf16* __restrict__ ktb,
                                                __bf16* __restrict__ vb) {
  int wave = threadIdx.x >> 5, lane = threadIdx.x & 31;
  int m0 = blockIdx.x * 128 + (wave >> 1) * 32;
  int n0 = blockIdx.y * 128 + (wave & 1) * 64;
  v8f acc[2][4] = {};
  for (int k = 0; k < C_DIM; k += 32) {
    v16bf a0 = frag_a(W, C_DIM, m0, k);
    v16bf a1 = frag_a(W, C_DIM, m0 + 16, k);
    v16bf bfr[4];
#pragma unroll
    for (int t = 0; t < 4; ++t) bfr[t] = frag_b(H, N_DIM, k, n0 + 16 * t);
#pragma unroll
    for (int t = 0; t < 4; ++t) {
      acc[0][t] = wmma_bf16(a0, bfr[t], acc[0][t]);
      acc[1][t] = wmma_bf16(a1, bfr[t], acc[1][t]);
    }
  }
  int col = lane & 15, rowh = (lane >> 4) << 3;
#pragma unroll
  for (int i = 0; i < 2; ++i)
#pragma unroll
    for (int t = 0; t < 4; ++t)
#pragma unroll
      for (int r = 0; r < 8; ++r) {
        int o = m0 + 16 * i + rowh + r;
        int n = n0 + 16 * t + col;
        float v = acc[i][t][r] + bias[o];
        if (o < C_DIM) {                       // Q: [head][n][d], pre-scaled
          int head = o >> 6, dk = o & 63;
          qb[((size_t)(head << 12) + n) * D_HEAD + dk] = (__bf16)(v * 0.125f);
        } else if (o < 2 * C_DIM) {            // K transposed: [head][d][n]
          int oc = o - C_DIM;
          int head = oc >> 6, dk = oc & 63;
          ktb[(size_t)((head << 6) + dk) * N_DIM + n] = (__bf16)v;
        } else {                               // V: [head][n][d]
          int oc = o - 2 * C_DIM;
          int head = oc >> 6, dk = oc & 63;
          vb[((size_t)(head << 12) + n) * D_HEAD + dk] = (__bf16)v;
        }
      }
}

// ---------------- Flash attention: 1 wave = 16 queries, full d=64 -----------

__global__ __launch_bounds__(128) void flash_attn(const __bf16* __restrict__ qg,
                                                  const __bf16* __restrict__ ktg,
                                                  const __bf16* __restrict__ vg,
                                                  __bf16* __restrict__ og) {
  __shared__ float  s_lds[4][16][64];
  __shared__ __bf16 p_lds[4][16][72];
  int head = blockIdx.y;
  int wave = threadIdx.x >> 5, lane = threadIdx.x & 31;
  int q0 = blockIdx.x * 64 + wave * 16;
  const __bf16* Q  = qg  + ((size_t)head << 12) * D_HEAD;   // [4096][64]
  const __bf16* KT = ktg + ((size_t)head << 6) * N_DIM;     // [64][4096]
  const __bf16* V  = vg  + ((size_t)head << 12) * D_HEAD;   // [4096][64]

  v16bf qf0 = frag_a(Q, D_HEAD, q0, 0);
  v16bf qf1 = frag_a(Q, D_HEAD, q0, 32);
  v8f o[4] = {};
  float m_run = -3.0e38f, l_run = 0.f;
  int row = lane & 15, half = lane >> 4;
  const int c0 = half << 5;

  for (int j0 = 0; j0 < N_DIM; j0 += 64) {
    // S = (Q*scale) . K^T for 16 queries x 64 keys
    v8f s[4] = {};
#pragma unroll
    for (int t = 0; t < 4; ++t) {
      s[t] = wmma_bf16(qf0, frag_b(KT, N_DIM, 0,  j0 + 16 * t), s[t]);
      s[t] = wmma_bf16(qf1, frag_b(KT, N_DIM, 32, j0 + 16 * t), s[t]);
    }
#pragma unroll
    for (int t = 0; t < 4; ++t)
#pragma unroll
      for (int r = 0; r < 8; ++r)
        s_lds[wave][r + (half << 3)][16 * t + row] = s[t][r];
    __syncthreads();

    // Online softmax: 2 lanes per row, 32 keys each.
    float mx = -3.0e38f;
#pragma unroll
    for (int jj = 0; jj < 32; ++jj) mx = fmaxf(mx, s_lds[wave][row][c0 + jj]);
    mx = fmaxf(mx, __shfl_xor(mx, 16));
    float mnew  = fmaxf(m_run, mx);
    float alpha = __expf(m_run - mnew);
    float ssum = 0.f;
#pragma unroll
    for (int jj = 0; jj < 32; ++jj) {
      float e = __expf(s_lds[wave][row][c0 + jj] - mnew);
      p_lds[wave][row][c0 + jj] = (__bf16)e;
      ssum += e;
    }
    ssum += __shfl_xor(ssum, 16);
    l_run = l_run * alpha + ssum;
    m_run = mnew;
    __syncthreads();

    // Rescale O accumulators per row (broadcast alpha via shuffle), O += P.V
#pragma unroll
    for (int r = 0; r < 8; ++r) {
      float ar = __shfl(alpha, (half << 3) + r);
#pragma unroll
      for (int t = 0; t < 4; ++t) o[t][r] *= ar;
    }
    v16bf pf0 = frag_a(&p_lds[wave][0][0], 72, 0, 0);
    v16bf pf1 = frag_a(&p_lds[wave][0][0], 72, 0, 32);
#pragma unroll
    for (int t = 0; t < 4; ++t) {
      o[t] = wmma_bf16(pf0, frag_b(V, D_HEAD, j0,      16 * t), o[t]);
      o[t] = wmma_bf16(pf1, frag_b(V, D_HEAD, j0 + 32, 16 * t), o[t]);
    }
  }

  float linv = 1.0f / l_run;
#pragma unroll
  for (int r = 0; r < 8; ++r) {
    float lr = __shfl(linv, (half << 3) + r);
    int qn = q0 + (half << 3) + r;
#pragma unroll
    for (int t = 0; t < 4; ++t) {
      int dk = 16 * t + row;
      og[(size_t)((head << 6) + dk) * N_DIM + qn] = (__bf16)(o[t][r] * lr);
    }
  }
}

// ------------- Proj GEMM [512,512]x[512,4096] + bias + residual -------------

__global__ __launch_bounds__(256) void proj_gemm(const __bf16* __restrict__ W,
                                                 const __bf16* __restrict__ O,
                                                 const float* __restrict__ bias,
                                                 const float* __restrict__ x,
                                                 float* __restrict__ out) {
  int wave = threadIdx.x >> 5, lane = threadIdx.x & 31;
  int m0 = blockIdx.x * 128 + (wave >> 1) * 32;
  int n0 = blockIdx.y * 128 + (wave & 1) * 64;
  v8f acc[2][4] = {};
  for (int k = 0; k < C_DIM; k += 32) {
    v16bf a0 = frag_a(W, C_DIM, m0, k);
    v16bf a1 = frag_a(W, C_DIM, m0 + 16, k);
    v16bf bfr[4];
#pragma unroll
    for (int t = 0; t < 4; ++t) bfr[t] = frag_b(O, N_DIM, k, n0 + 16 * t);
#pragma unroll
    for (int t = 0; t < 4; ++t) {
      acc[0][t] = wmma_bf16(a0, bfr[t], acc[0][t]);
      acc[1][t] = wmma_bf16(a1, bfr[t], acc[1][t]);
    }
  }
  int col = lane & 15, rowh = (lane >> 4) << 3;
#pragma unroll
  for (int i = 0; i < 2; ++i)
#pragma unroll
    for (int t = 0; t < 4; ++t)
#pragma unroll
      for (int r = 0; r < 8; ++r) {
        int oc = m0 + 16 * i + rowh + r;
        int n  = n0 + 16 * t + col;
        size_t idx = (size_t)oc * N_DIM + n;
        out[idx] = acc[i][t][r] + bias[oc] + x[idx];
      }
}

// ---------------------------------------------------------------------------

extern "C" void kernel_launch(void* const* d_in, const int* in_sizes, int n_in,
                              void* d_out, int out_size, void* d_ws, size_t ws_size,
                              hipStream_t stream) {
  const float* x      = (const float*)d_in[0];
  const float* norm_w = (const float*)d_in[1];
  const float* norm_b = (const float*)d_in[2];
  const float* qkv_w  = (const float*)d_in[3];
  const float* qkv_b  = (const float*)d_in[4];
  const float* proj_w = (const float*)d_in[5];
  const float* proj_b = (const float*)d_in[6];
  float* out = (float*)d_out;

  char* ws = (char*)d_ws;
  size_t off = 0;
  auto alloc = [&](size_t bytes) -> void* {
    off = (off + 255) & ~(size_t)255;
    void* p = ws + off;
    off += bytes;
    return p;
  };
  float*  partials = (float*)alloc(128 * 2 * sizeof(float));
  float*  stats    = (float*)alloc(GROUPS * 2 * sizeof(float));
  __bf16* h_bf  = (__bf16*)alloc((size_t)C_DIM * N_DIM * 2);       // 4 MB
  __bf16* qw_bf = (__bf16*)alloc((size_t)3 * C_DIM * C_DIM * 2);   // 1.5 MB
  __bf16* pw_bf = (__bf16*)alloc((size_t)C_DIM * C_DIM * 2);       // 0.5 MB
  __bf16* q_bf  = (__bf16*)alloc((size_t)C_DIM * N_DIM * 2);       // 4 MB
  __bf16* kt_bf = (__bf16*)alloc((size_t)C_DIM * N_DIM * 2);       // 4 MB
  __bf16* v_bf  = (__bf16*)alloc((size_t)C_DIM * N_DIM * 2);       // 4 MB
  __bf16* o_bf  = (__bf16*)alloc((size_t)C_DIM * N_DIM * 2);       // 4 MB
  (void)ws_size; (void)in_sizes; (void)n_in; (void)out_size;

  gn_partial<<<128, 256, 0, stream>>>(x, partials);
  gn_finalize<<<1, 32, 0, stream>>>(partials, stats);
  gn_apply<<<2048, 256, 0, stream>>>(x, norm_w, norm_b, stats, h_bf);
  cvt_bf16_k<<<768, 256, 0, stream>>>(qkv_w, qw_bf, 196608);
  cvt_bf16_k<<<256, 256, 0, stream>>>(proj_w, pw_bf, 65536);
  qkv_gemm<<<dim3(12, 32), 256, 0, stream>>>(qw_bf, h_bf, qkv_b, q_bf, kt_bf, v_bf);
  flash_attn<<<dim3(64, HEADS), 128, 0, stream>>>(q_bf, kt_bf, v_bf, o_bf);
  proj_gemm<<<dim3(4, 32), 256, 0, stream>>>(pw_bf, o_bf, proj_b, x, out);
}
